// MultiHeadAttentionQuantum_65481071404695
// MI455X (gfx1250) — matrix operand order repaired and checked by reading, not compiled, not run on hardware
//
#include <hip/hip_runtime.h>
#include <math.h>

typedef __attribute__((ext_vector_type(16))) _Float16 v16h;
typedef __attribute__((ext_vector_type(8)))  _Float16 v8h;
typedef __attribute__((ext_vector_type(8)))  float    v8f;

#define BATCH 8
#define SEQ   2048
#define EMB   768
#define SCALE 0.35355339059327373f   // 1/sqrt(8)

union ABu { v16h v; v8h h[2]; };

// ---------------------------------------------------------------------------
// K0a: q = cos(x + theta) -> f16, plus transposed copy qT ([b][e][s]).
// LDS-tiled so both q and qT writes are coalesced.
// ---------------------------------------------------------------------------
__global__ __launch_bounds__(256)
void qgen_kernel(const float* __restrict__ x, const float* __restrict__ theta,
                 _Float16* __restrict__ q, _Float16* __restrict__ qT) {
    __shared__ _Float16 tile[64][72];           // pad to dodge bank conflicts
    const int tid = threadIdx.x;
    const int b   = blockIdx.z;
    const int s0  = blockIdx.x * 64;
    const int e0  = blockIdx.y * 64;
    const int eL  = tid & 63;
    const int sB  = tid >> 6;                    // 0..3
    const float th = theta[(e0 + eL) & 7];       // EMB % 8 == 0

    const float*  xb = x + (size_t)b * SEQ * EMB;
    _Float16*     qb = q + (size_t)b * SEQ * EMB;
    #pragma unroll
    for (int i = 0; i < 16; ++i) {
        const int sL = sB + i * 4;
        const float v = __cosf(xb[(size_t)(s0 + sL) * EMB + e0 + eL] + th);
        const _Float16 hv = (_Float16)v;
        qb[(size_t)(s0 + sL) * EMB + e0 + eL] = hv;
        tile[sL][eL] = hv;
    }
    __syncthreads();
    _Float16* qTb = qT + (size_t)b * EMB * SEQ;
    const int sOut = tid & 63;
    #pragma unroll
    for (int i = 0; i < 16; ++i) {
        const int eR = sB + i * 4;
        qTb[(size_t)(e0 + eR) * SEQ + s0 + sOut] = tile[sOut][eR];
    }
}

// ---------------------------------------------------------------------------
// K0b: W_combine f32 -> f16
// ---------------------------------------------------------------------------
__global__ void wconv_kernel(const float* __restrict__ W, _Float16* __restrict__ Wh) {
    const int i = blockIdx.x * blockDim.x + threadIdx.x;
    if (i < EMB * EMB) Wh[i] = (_Float16)W[i];
}

// ---------------------------------------------------------------------------
// K1: fused softmax attention for one (batch, 16-row query tile) per WG.
// 4 waves / WG. Phase 1: per-row (max, sum-exp) via WMMA QK^T + shuffle
// reductions. Phase 2: normalized P tiles staged in LDS (A-operand layout),
// P @ V accumulated per-wave over a 192-wide output slice.
// ---------------------------------------------------------------------------
__global__ __launch_bounds__(128)
void attn_kernel(const _Float16* __restrict__ q,
                 const _Float16* __restrict__ qT,
                 _Float16* __restrict__ attn) {
    __shared__ float    mW[4][16], lW[4][16];
    __shared__ float    mTot[16], lInv[16];
    __shared__ _Float16 Pbuf[4][16 * 32];        // 4 blocks of P (16x32), row-major

    const int tid  = threadIdx.x;
    const int lane = tid & 31;
    const int wave = tid >> 5;
    const int b    = blockIdx.x >> 7;            // 128 row tiles per batch
    const int s0   = (blockIdx.x & 127) * 16;

    const _Float16* qb  = q  + (size_t)b * SEQ * EMB;
    const _Float16* qTb = qT + (size_t)b * EMB * SEQ;

    const int nlo = lane & 15;
    const int hi8 = (lane >> 4) * 8;             // C-tile row offset: 0 or 8
    const int kbA = (lane < 16) ? 0 : 16;        // A-operand K base (contiguous k = 16h + s)
    const int kbB = (lane < 16) ? 0 : 16;        // B-operand K base (contiguous k = 16h + s)

    const _Float16* qrowA = qb + (size_t)(s0 + nlo) * EMB;   // Q row for this lane

    // ---------------- phase 1: (m, l) per row ----------------
    float m[8], l[8];
    #pragma unroll
    for (int r = 0; r < 8; ++r) { m[r] = -__builtin_inff(); l[r] = 0.0f; }

    for (int t0 = wave * 16; t0 < SEQ; t0 += 64) {
        v8f acc = {};
        const _Float16* qrowB = qb + (size_t)(t0 + nlo) * EMB;
        for (int kc = 0; kc < EMB; kc += 32) {
            ABu a, bm;
            a.h[0]  = *(const v8h*)(qrowA + kc + kbA);
            a.h[1]  = *(const v8h*)(qrowA + kc + kbA + 8);
            bm.h[0] = *(const v8h*)(qrowB + kc + kbB);
            bm.h[1] = *(const v8h*)(qrowB + kc + kbB + 8);
            acc = __builtin_amdgcn_wmma_f32_16x16x32_f16(false, a.v, false, bm.v,
                                                         (short)0, acc, false, false);
        }
        #pragma unroll
        for (int r = 0; r < 8; ++r) {
            float s = acc[r] * SCALE;
            float tmax = s;
            tmax = fmaxf(tmax, __shfl_xor(tmax, 1));
            tmax = fmaxf(tmax, __shfl_xor(tmax, 2));
            tmax = fmaxf(tmax, __shfl_xor(tmax, 4));
            tmax = fmaxf(tmax, __shfl_xor(tmax, 8));
            const float nm = fmaxf(m[r], tmax);
            float p = __expf(s - nm);
            p += __shfl_xor(p, 1);
            p += __shfl_xor(p, 2);
            p += __shfl_xor(p, 4);
            p += __shfl_xor(p, 8);
            l[r] = l[r] * __expf(m[r] - nm) + p;
            m[r] = nm;
        }
    }
    if (nlo == 0) {
        #pragma unroll
        for (int r = 0; r < 8; ++r) { mW[wave][r + hi8] = m[r]; lW[wave][r + hi8] = l[r]; }
    }
    __syncthreads();
    if (tid < 16) {
        const float mt = fmaxf(fmaxf(mW[0][tid], mW[1][tid]),
                               fmaxf(mW[2][tid], mW[3][tid]));
        float lt = 0.0f;
        #pragma unroll
        for (int w = 0; w < 4; ++w) lt += lW[w][tid] * __expf(mW[w][tid] - mt);
        mTot[tid] = mt;
        lInv[tid] = 1.0f / lt;                   // lt >= 1 always (exp(0) term)
    }
    __syncthreads();

    float mt[8], li[8];
    #pragma unroll
    for (int r = 0; r < 8; ++r) { mt[r] = mTot[r + hi8]; li[r] = lInv[r + hi8]; }

    // ---------------- phase 2: O = P_hat @ V ----------------
    v8f O[12];
    const v8f vzero = {};
    #pragma unroll
    for (int n = 0; n < 12; ++n) O[n] = vzero;

    const int e0 = wave * 192;                   // this wave's output slice
    for (int tb = 0; tb < SEQ; tb += 128) {
        const int t0 = tb + wave * 32;           // this wave builds P for 32 cols
        #pragma unroll
        for (int half = 0; half < 2; ++half) {
            v8f acc = {};
            const _Float16* qrowB = qb + (size_t)(t0 + half * 16 + nlo) * EMB;
            for (int kc = 0; kc < EMB; kc += 32) {
                ABu a, bm;
                a.h[0]  = *(const v8h*)(qrowA + kc + kbA);
                a.h[1]  = *(const v8h*)(qrowA + kc + kbA + 8);
                bm.h[0] = *(const v8h*)(qrowB + kc + kbB);
                bm.h[1] = *(const v8h*)(qrowB + kc + kbB + 8);
                acc = __builtin_amdgcn_wmma_f32_16x16x32_f16(false, a.v, false, bm.v,
                                                             (short)0, acc, false, false);
            }
            #pragma unroll
            for (int r = 0; r < 8; ++r) {
                const float p = __expf(acc[r] * SCALE - mt[r]) * li[r];
                Pbuf[wave][(r + hi8) * 32 + half * 16 + nlo] = (_Float16)p;
            }
        }
        __syncthreads();
        #pragma unroll
        for (int blk = 0; blk < 4; ++blk) {
            ABu a;
            const _Float16* pb = &Pbuf[blk][0];
            a.h[0] = *(const v8h*)(pb + nlo * 32 + kbA);
            a.h[1] = *(const v8h*)(pb + nlo * 32 + kbA + 8);
            const int tk = tb + blk * 32;
            #pragma unroll
            for (int n = 0; n < 12; ++n) {
                ABu bm;
                const _Float16* vcol = qTb + (size_t)(e0 + n * 16 + nlo) * SEQ + tk;
                bm.h[0] = *(const v8h*)(vcol + kbB);
                bm.h[1] = *(const v8h*)(vcol + kbB + 8);
                O[n] = __builtin_amdgcn_wmma_f32_16x16x32_f16(false, a.v, false, bm.v,
                                                              (short)0, O[n], false, false);
            }
        }
        __syncthreads();
    }

    _Float16* ab = attn + (size_t)b * SEQ * EMB;
    #pragma unroll
    for (int n = 0; n < 12; ++n)
        #pragma unroll
        for (int r = 0; r < 8; ++r)
            ab[(size_t)(s0 + r + hi8) * EMB + e0 + n * 16 + nlo] = (_Float16)O[n][r];
}

// ---------------------------------------------------------------------------
// K2: out = attn @ W^T (f16 x f16 -> f32), 16 rows x 64 cols per wave.
// ---------------------------------------------------------------------------
__global__ __launch_bounds__(128)
void epi_kernel(const _Float16* __restrict__ attn,
                const _Float16* __restrict__ Wh,
                float* __restrict__ out) {
    const int tid  = threadIdx.x;
    const int lane = tid & 31;
    const int wave = tid >> 5;
    const int job  = blockIdx.x * 4 + wave;      // 12288 jobs total
    const int rt   = job / 12;                   // 1024 row tiles (b*S flattened)
    const int cg   = job % 12;                   // 12 column groups of 64
    const int nlo  = lane & 15;
    const int hi8  = (lane >> 4) * 8;
    const int kbA  = (lane < 16) ? 0 : 16;
    const int kbB  = (lane < 16) ? 0 : 16;

    const _Float16* arow = attn + (size_t)(rt * 16 + nlo) * EMB;
    v8f acc[4];
    const v8f vzero = {};
    #pragma unroll
    for (int n = 0; n < 4; ++n) acc[n] = vzero;

    for (int kc = 0; kc < EMB; kc += 32) {
        ABu a;
        a.h[0] = *(const v8h*)(arow + kc + kbA);
        a.h[1] = *(const v8h*)(arow + kc + kbA + 8);
        #pragma unroll
        for (int n = 0; n < 4; ++n) {
            ABu bm;
            const _Float16* wrow = Wh + (size_t)(cg * 64 + n * 16 + nlo) * EMB + kc;
            bm.h[0] = *(const v8h*)(wrow + kbB);
            bm.h[1] = *(const v8h*)(wrow + kbB + 8);
            acc[n] = __builtin_amdgcn_wmma_f32_16x16x32_f16(false, a.v, false, bm.v,
                                                            (short)0, acc[n], false, false);
        }
    }
    #pragma unroll
    for (int n = 0; n < 4; ++n)
        #pragma unroll
        for (int r = 0; r < 8; ++r)
            out[(size_t)(rt * 16 + r + hi8) * EMB + cg * 64 + n * 16 + nlo] = acc[n][r];
}

// ---------------------------------------------------------------------------
// Workspace layout (bytes):
//   [0,              25165824)  q   f16  [b][s][e]
//   [25165824,       50331648)  qT  f16  [b][e][s]
//   [50331648,       75497472)  attn f16 [b][s][e]
//   [75497472,       76677120)  Wh  f16  [o][e]
// Requires ws_size >= ~76.7 MB.
// ---------------------------------------------------------------------------
extern "C" void kernel_launch(void* const* d_in, const int* in_sizes, int n_in,
                              void* d_out, int out_size, void* d_ws, size_t ws_size,
                              hipStream_t stream) {
    (void)in_sizes; (void)n_in; (void)out_size; (void)ws_size;
    const float* x     = (const float*)d_in[0];
    const float* theta = (const float*)d_in[1];
    const float* W     = (const float*)d_in[2];
    float*       out   = (float*)d_out;

    char* ws = (char*)d_ws;
    const size_t qBytes = (size_t)BATCH * SEQ * EMB * sizeof(_Float16);
    _Float16* q  = (_Float16*)(ws);
    _Float16* qT = (_Float16*)(ws + qBytes);
    _Float16* at = (_Float16*)(ws + 2 * qBytes);
    _Float16* Wh = (_Float16*)(ws + 3 * qBytes);

    dim3 g0(SEQ / 64, EMB / 64, BATCH);
    qgen_kernel<<<g0, 256, 0, stream>>>(x, theta, q, qT);
    wconv_kernel<<<(EMB * EMB + 255) / 256, 256, 0, stream>>>(W, Wh);
    attn_kernel<<<BATCH * (SEQ / 16), 128, 0, stream>>>(q, qT, at);
    epi_kernel<<<(1024 * 12) / 4, 128, 0, stream>>>(at, Wh, out);
}
